// SpikingExpertGroup_25262997636016
// MI455X (gfx1250) — compile-verified, matrix operand order checked by reading, not run on hardware
//
#include <hip/hip_runtime.h>
#include <hip/hip_bf16.h>
#include <cstdint>

typedef __attribute__((ext_vector_type(16))) __bf16 bf16x16;
typedef __attribute__((ext_vector_type(8)))  float  f32x8;

// Problem constants (fixed by setup_inputs)
#define TT 4
#define NN 8192
#define DD 1024
#define EE 8
#define EF 512

#define WMMA_BF16(A, B, C) \
    __builtin_amdgcn_wmma_f32_16x16x32_bf16(false, (A), false, (B), (short)0, (C), false, false)

// ---------------- helpers ----------------
__global__ void k_zero4(float4* __restrict__ p, int n4) {
    int i = blockIdx.x * blockDim.x + threadIdx.x;
    if (i < n4) p[i] = make_float4(0.f, 0.f, 0.f, 0.f);
}

__device__ __forceinline__ unsigned short f2bf(float f) {
    unsigned int u = __float_as_uint(f);
    return (unsigned short)((u + 0x7FFFu + ((u >> 16) & 1u)) >> 16); // RNE
}

// generic convert (weights)
__global__ void k_f2bf(const float* __restrict__ s, unsigned short* __restrict__ d, int n) {
    int i = blockIdx.x * blockDim.x + threadIdx.x;
    if (i < n) d[i] = f2bf(s[i]);
}

// convert x with [T][N][D] -> [N][T][D] transpose so all 4 time-steps of a token
// sit within 24-bit-immediate reach of one base address. N*D = 2^23, D = 2^10.
__global__ void k_f2bf_x(const float* __restrict__ s, unsigned short* __restrict__ d) {
    int i = blockIdx.x * blockDim.x + threadIdx.x;           // flat src index (t,n,d)
    int t = i >> 23;
    int n = (i >> 10) & (NN - 1);
    int dd = i & (DD - 1);
    d[(n * TT + t) * DD + dd] = f2bf(s[i]);
}

__global__ void k_scatter_w(const int* __restrict__ idx, const float* __restrict__ wts,
                            float* __restrict__ Wd) {
    int n = blockIdx.x * blockDim.x + threadIdx.x;
    if (n < NN) {
        atomicAdd(&Wd[n * EE + idx[n * 2 + 0]], wts[n * 2 + 0]);
        atomicAdd(&Wd[n * EE + idx[n * 2 + 1]], wts[n * 2 + 1]);
    }
}

__global__ void k_build_list(const float* __restrict__ Wd, int* __restrict__ cnt,
                             int* __restrict__ lst) {
    int id = blockIdx.x * blockDim.x + threadIdx.x;
    if (id < NN * EE && Wd[id] != 0.0f) {
        int e = id & (EE - 1);
        int n = id >> 3;
        int pos = atomicAdd(&cnt[e], 1);
        lst[e * NN + pos] = n;
    }
}

// LIF over 4 time-step accumulators -> bf16 spikes into LDS column
__device__ __forceinline__ void lif_spikes(unsigned short* __restrict__ H, int m_off, int fcol,
                                           const f32x8 a0, const f32x8 a1,
                                           const f32x8 a2, const f32x8 a3) {
    #pragma unroll
    for (int i = 0; i < 8; ++i) {
        unsigned short* hp = &H[(m_off + i) * EF + fcol];
        unsigned short s;
        float mem = a0[i];
        s = 0; if (mem > 1.0f) { s = 0x3F80u; mem -= 1.0f; } hp[0 * 16 * EF] = s;
        mem = 0.9f * mem + a1[i];
        s = 0; if (mem > 1.0f) { s = 0x3F80u; mem -= 1.0f; } hp[1 * 16 * EF] = s;
        mem = 0.9f * mem + a2[i];
        s = 0; if (mem > 1.0f) { s = 0x3F80u; mem -= 1.0f; } hp[2 * 16 * EF] = s;
        mem = 0.9f * mem + a3[i];
        s = 0; if (mem > 1.0f) { s = 0x3F80u; mem -= 1.0f; } hp[3 * 16 * EF] = s;
    }
}

// ---------------- fused spiking-MoE expert kernel ----------------
// grid = EE * (NN/16) blocks; block = 256 threads (8 wave32)
__global__ __launch_bounds__(256) void k_moe(
    const unsigned short* __restrict__ xb,   // bf16 x  [N][T][D]  (transposed)
    const unsigned short* __restrict__ ub,   // bf16 up_w [E][EF][D]
    const unsigned short* __restrict__ db,   // bf16 down_w [E][D][EF]
    const float* __restrict__ Wd,            // [N][E] combine weights
    const int*   __restrict__ cnt,           // [E] routed counts
    const int*   __restrict__ lst,           // [E][N] routed token ids
    float* __restrict__ out)                 // f32 [T][N][D], pre-zeroed
{
    __shared__ alignas(32) unsigned short H[TT * 16 * EF]; // 64 KB spike buffer

    const int bx   = blockIdx.x;
    const int e    = bx >> 9;          // / (NN/16)
    const int tile = bx & 511;
    const int c    = cnt[e];
    if (tile * 16 >= c) return;        // early-exit for unused tiles

    const int tid   = threadIdx.x;
    const int wave  = tid >> 5;
    const int lane  = tid & 31;
    const int col   = lane & 15;       // M row (A) / N col (B,C)
    const int hi    = lane >> 4;       // half-wave -> K range / M+8
    const int khalf = hi * 16;

    // each lane holds token-id/weight for row m == col (both half-waves)
    int slot = tile * 16 + col;
    int tok; float wgt;
    if (slot < c) { tok = lst[e * NN + slot]; wgt = Wd[tok * EE + e]; }
    else          { tok = lst[e * NN];        wgt = 0.0f; }           // padded: weight 0

    const unsigned short* aX = xb + (size_t)tok * (TT * DD);  // t strides = 2 KB immediates

    // ---------- phase 1: up-proj + LIF -> spikes in LDS ----------
    // each wave: 4 f-tiles as 2 pairs (2 B operands share 4 A operands -> 6 loads : 8 WMMA)
    #pragma unroll
    for (int j = 0; j < 2; ++j) {
        const int fbase = (wave * 4 + j * 2) * 16;            // pair covers fbase, fbase+16
        const unsigned short* bU = ub + ((size_t)(e * EF + fbase + col)) * DD;

        f32x8 p0 = {}, p1 = {}, p2 = {}, p3 = {};             // tile 0, t = 0..3
        f32x8 q0 = {}, q1 = {}, q2 = {}, q3 = {};             // tile 1, t = 0..3
        for (int kt = 0; kt < DD / 32; ++kt) {
            const int kb = kt * 32 + khalf;
            bf16x16 B0 = *(const bf16x16*)(bU + kb);
            bf16x16 B1 = *(const bf16x16*)(bU + 16 * DD + kb);   // +32 KB immediate
            bf16x16 A0 = *(const bf16x16*)(aX + 0 * DD + kb);
            bf16x16 A1 = *(const bf16x16*)(aX + 1 * DD + kb);
            bf16x16 A2 = *(const bf16x16*)(aX + 2 * DD + kb);
            bf16x16 A3 = *(const bf16x16*)(aX + 3 * DD + kb);
            p0 = WMMA_BF16(A0, B0, p0);  q0 = WMMA_BF16(A0, B1, q0);
            p1 = WMMA_BF16(A1, B0, p1);  q1 = WMMA_BF16(A1, B1, q1);
            p2 = WMMA_BF16(A2, B0, p2);  q2 = WMMA_BF16(A2, B1, q2);
            p3 = WMMA_BF16(A3, B0, p3);  q3 = WMMA_BF16(A3, B1, q3);
        }
        lif_spikes(H, 8 * hi, fbase + col,      p0, p1, p2, p3);
        lif_spikes(H, 8 * hi, fbase + 16 + col, q0, q1, q2, q3);
    }
    __syncthreads();

    // ---------- phase 2: down-proj + LIF + weighted combine ----------
    // each wave: 8 d-tiles as 4 pairs
    #pragma unroll
    for (int j = 0; j < 4; ++j) {
        const int dbase = (wave * 8 + j * 2) * 16;            // pair covers dbase, dbase+16
        const unsigned short* bD = db + ((size_t)(e * DD + dbase + col)) * EF;

        f32x8 p0 = {}, p1 = {}, p2 = {}, p3 = {};
        f32x8 q0 = {}, q1 = {}, q2 = {}, q3 = {};
        for (int kt = 0; kt < EF / 32; ++kt) {
            const int kb = kt * 32 + khalf;
            bf16x16 B0 = *(const bf16x16*)(bD + kb);
            bf16x16 B1 = *(const bf16x16*)(bD + 16 * EF + kb);   // +16 KB immediate
            bf16x16 A0 = *(const bf16x16*)(&H[(0 * 16 + col) * EF + kb]);
            bf16x16 A1 = *(const bf16x16*)(&H[(1 * 16 + col) * EF + kb]);
            bf16x16 A2 = *(const bf16x16*)(&H[(2 * 16 + col) * EF + kb]);
            bf16x16 A3 = *(const bf16x16*)(&H[(3 * 16 + col) * EF + kb]);
            p0 = WMMA_BF16(A0, B0, p0);  q0 = WMMA_BF16(A0, B1, q0);
            p1 = WMMA_BF16(A1, B0, p1);  q1 = WMMA_BF16(A1, B1, q1);
            p2 = WMMA_BF16(A2, B0, p2);  q2 = WMMA_BF16(A2, B1, q2);
            p3 = WMMA_BF16(A3, B0, p3);  q3 = WMMA_BF16(A3, B1, q3);
        }
        #pragma unroll
        for (int i = 0; i < 8; ++i) {
            const int m    = i + 8 * hi;
            const int tkm  = __shfl(tok, m);
            const float wm = __shfl(wgt, m);
            float* op0 = out + (size_t)tkm * DD + dbase + col;       // tile 0 column
            float* op1 = op0 + 16;                                    // tile 1 column
            // tile 0 LIF chain
            float mem = p0[i];
            if (mem > 1.0f) { mem -= 1.0f; if (wm != 0.0f) atomicAdd(op0, wm); }
            mem = 0.9f * mem + p1[i];
            if (mem > 1.0f) { mem -= 1.0f; if (wm != 0.0f) atomicAdd(op0 + (size_t)1 * NN * DD, wm); }
            mem = 0.9f * mem + p2[i];
            if (mem > 1.0f) { mem -= 1.0f; if (wm != 0.0f) atomicAdd(op0 + (size_t)2 * NN * DD, wm); }
            mem = 0.9f * mem + p3[i];
            if (mem > 1.0f) { mem -= 1.0f; if (wm != 0.0f) atomicAdd(op0 + (size_t)3 * NN * DD, wm); }
            // tile 1 LIF chain
            mem = q0[i];
            if (mem > 1.0f) { mem -= 1.0f; if (wm != 0.0f) atomicAdd(op1, wm); }
            mem = 0.9f * mem + q1[i];
            if (mem > 1.0f) { mem -= 1.0f; if (wm != 0.0f) atomicAdd(op1 + (size_t)1 * NN * DD, wm); }
            mem = 0.9f * mem + q2[i];
            if (mem > 1.0f) { mem -= 1.0f; if (wm != 0.0f) atomicAdd(op1 + (size_t)2 * NN * DD, wm); }
            mem = 0.9f * mem + q3[i];
            if (mem > 1.0f) { mem -= 1.0f; if (wm != 0.0f) atomicAdd(op1 + (size_t)3 * NN * DD, wm); }
        }
    }
}

// ---------------- launch ----------------
extern "C" void kernel_launch(void* const* d_in, const int* in_sizes, int n_in,
                              void* d_out, int out_size, void* d_ws, size_t ws_size,
                              hipStream_t stream) {
    const float* x    = (const float*)d_in[0];   // [T][N][D] f32
    const int*   eidx = (const int*)  d_in[1];   // [N][2]
    const float* ewt  = (const float*)d_in[2];   // [N][2]
    const float* upw  = (const float*)d_in[3];   // [E][EF][D]
    const float* dnw  = (const float*)d_in[4];   // [E][D][EF]
    float* out = (float*)d_out;

    uint8_t* ws = (uint8_t*)d_ws;
    unsigned short* xb = (unsigned short*)(ws + 0);           // 67,108,864 B  [N][T][D]
    unsigned short* ub = (unsigned short*)(ws + 67108864);    //  8,388,608 B
    unsigned short* db = (unsigned short*)(ws + 75497472);    //  8,388,608 B
    float*          Wd = (float*)        (ws + 83886080);     //    262,144 B
    int*            ct = (int*)          (ws + 84148224);     //        256 B
    int*            ls = (int*)          (ws + 84148480);     //    262,144 B

    // 1) zero output (T*N*D floats) and routing scratch (Wd+cnt contiguous)
    k_zero4<<<32768, 256, 0, stream>>>((float4*)out, (TT * NN * DD) / 4);
    k_zero4<<<65, 256, 0, stream>>>((float4*)Wd, (NN * EE + 64) / 4);

    // 2) fp32 -> bf16 conversions (x transposed to [N][T][D])
    k_f2bf_x<<<(TT * NN * DD) / 256, 256, 0, stream>>>(x, xb);
    k_f2bf<<<(EE * EF * DD) / 256, 256, 0, stream>>>(upw, ub, EE * EF * DD);
    k_f2bf<<<(EE * DD * EF) / 256, 256, 0, stream>>>(dnw, db, EE * DD * EF);

    // 3) routing: dense combine weights, then per-expert compacted token lists
    k_scatter_w<<<NN / 256, 256, 0, stream>>>(eidx, ewt, Wd);
    k_build_list<<<(NN * EE) / 256, 256, 0, stream>>>(Wd, ct, ls);

    // 4) fused expert kernel: 8 experts x 512 token-tiles, early-exit beyond routed counts
    k_moe<<<EE * (NN / 16), 256, 0, stream>>>(xb, ub, db, Wd, ct, ls, out);
}
